// FocusedLinearAttention_6734508720660
// MI455X (gfx1250) — matrix operand order repaired and checked by reading, not compiled, not run on hardware
//
#include <hip/hip_runtime.h>
#include <cstddef>

typedef __attribute__((ext_vector_type(16))) _Float16 v16h;
typedef __attribute__((ext_vector_type(8)))  float    v8f;

#define GM 25088   // B*N
#define GK 768
#define GN 768
#define NS 196
#define HD 96
#define BH 1024    // B*H
#define WSZ ((size_t)GK * GN)   // one weight matrix

// ---------------------------------------------------------------------------
// CDNA5 async global->LDS staging (ASYNCcnt path), guarded so compile is safe
// Builtin signature (from hipcc diagnostic): (v4i AS1*, v4i AS3*, Imm, Imm)
// ---------------------------------------------------------------------------
#if defined(__has_builtin)
#if __has_builtin(__builtin_amdgcn_global_load_async_to_lds_b128)
#define HAVE_ASYNC_LDS 1
#endif
#endif

#ifdef HAVE_ASYNC_LDS
typedef int v4i_t __attribute__((vector_size(16)));
typedef __attribute__((address_space(1))) v4i_t gv4i_t;   // HIP __device__ AS
typedef __attribute__((address_space(3))) v4i_t lv4i_t;   // HIP __shared__ AS
__device__ __forceinline__ void async_cp16(const void* g, void* l) {
  // generic->AS1 is value-preserving; generic LDS pointer low 32 bits are the
  // LDS offset (flat->local is a truncation on AMDGPU).
  __builtin_amdgcn_global_load_async_to_lds_b128(
      (gv4i_t*)(unsigned long long)(size_t)g,
      (lv4i_t*)(unsigned int)(size_t)l, 0, 0);
}
__device__ __forceinline__ void wait_async0() {
#if __has_builtin(__builtin_amdgcn_s_wait_asynccnt)
  __builtin_amdgcn_s_wait_asynccnt(0);
#else
  asm volatile("s_wait_asynccnt 0x0" ::: "memory");
#endif
}
#endif

// ---------------------------------------------------------------------------
// Fragment loaders (all LDS strides multiples of 8 halves -> 16B aligned b128)
// A (16x32 f16 MxK): lane<16 -> row=lane, K {0..7,16..23}; lane>=16 -> {8..15,24..31}
// B (32x16 f16 KxN) staged transposed Bs[n][k]: lane<16 -> col=lane, K {0..7,16..23}; lane>=16 -> {8..15,24..31}
// ---------------------------------------------------------------------------
__device__ __forceinline__ v16h frag_a(const _Float16* row, int g0) {
  v16h r;
  ((uint4*)&r)[0] = *(const uint4*)(row + g0);
  ((uint4*)&r)[1] = *(const uint4*)(row + g0 + 16);
  return r;
}
__device__ __forceinline__ v16h frag_b(const _Float16* row, int g0) {
  v16h r;
  ((uint4*)&r)[0] = *(const uint4*)(row + g0);
  ((uint4*)&r)[1] = *(const uint4*)(row + g0 + 16);
  return r;
}

// ---------------------------------------------------------------------------
// x (f32) -> f16, 4 elems/thread
// ---------------------------------------------------------------------------
__global__ __launch_bounds__(256) void cvt_f16_kernel(
    const float* __restrict__ in, _Float16* __restrict__ out)
{
  size_t i = (size_t)blockIdx.x * 256 + threadIdx.x;
  float4 v = ((const float4*)in)[i];
  _Float16 h[4] = {(_Float16)v.x, (_Float16)v.y, (_Float16)v.z, (_Float16)v.w};
  *(uint2*)(out + i * 4) = *(uint2*)h;
}

// ---------------------------------------------------------------------------
// Transpose+convert 4 weights: W[k][n] f32 -> Wt[n][k] f16 (once, tiny)
// ---------------------------------------------------------------------------
__global__ __launch_bounds__(256) void wtrans_kernel(
    const float* __restrict__ Wq, const float* __restrict__ Wk,
    const float* __restrict__ Wv, const float* __restrict__ Wp,
    _Float16* __restrict__ wt)
{
  const float* W = blockIdx.z == 0 ? Wq : blockIdx.z == 1 ? Wk
                 : blockIdx.z == 2 ? Wv : Wp;
  _Float16* dst = wt + (size_t)blockIdx.z * WSZ;
  int k0 = blockIdx.y * 64, n0 = blockIdx.x * 64;
  for (int idx = threadIdx.x; idx < 4096; idx += 256) {
    int i = idx >> 6, j = idx & 63;
    dst[(size_t)(n0 + j) * GK + k0 + i] = (_Float16)W[(size_t)(k0 + i) * GN + n0 + j];
  }
}

// ---------------------------------------------------------------------------
// GEMM: (25088x768 f16) * Wt(768x768 f16, pre-transposed [n][k]) + bias.
// mode 0: f32 row-major out.  mode 1: f16 head-split out (bh,n,hd) [for V].
// Block 256 = 8 waves; tile 128x128; wave 64x32 (4x2 subtiles); K step 32.
// Staging = pure 16B copies -> async global->LDS when available.
// ---------------------------------------------------------------------------
__global__ __launch_bounds__(256) void gemm768h_kernel(
    const _Float16* __restrict__ A, const _Float16* __restrict__ Wt,
    const float* __restrict__ bias, float* __restrict__ outF32,
    _Float16* __restrict__ outSplit, int mode)
{
  __shared__ __align__(16) _Float16 As[128 * 48];
  __shared__ __align__(16) _Float16 Bs[128 * 48];
  const int tid = threadIdx.x;
  const int lane = tid & 31;
  const int laneLo = lane & 15;
  const int hi = (lane >> 4) & 1;
  const int wave = tid >> 5;
  const int wm = wave & 1;
  const int wn = wave >> 1;
  const int bm0 = blockIdx.y * 128;
  const int bn0 = blockIdx.x * 128;

  v8f acc[4][2];
  #pragma unroll
  for (int i = 0; i < 4; ++i)
    #pragma unroll
    for (int j = 0; j < 2; ++j)
      #pragma unroll
      for (int e = 0; e < 8; ++e) acc[i][j][e] = 0.f;

  for (int k0 = 0; k0 < GK; k0 += 32) {
    if (k0) __syncthreads();
    // 128 rows x 32 halves = 512 x 16B chunks per tile; 2 per thread per tile
#ifdef HAVE_ASYNC_LDS
    #pragma unroll
    for (int i = 0; i < 2; ++i) {
      int ch = tid + (i << 8);
      int r = ch >> 2, c4 = (ch & 3) << 3;
      async_cp16(A + (size_t)(bm0 + r) * GK + k0 + c4, &As[r * 48 + c4]);
      async_cp16(Wt + (size_t)(bn0 + r) * GK + k0 + c4, &Bs[r * 48 + c4]);
    }
#else
    #pragma unroll
    for (int i = 0; i < 2; ++i) {
      int ch = tid + (i << 8);
      int r = ch >> 2, c4 = (ch & 3) << 3;
      *(uint4*)&As[r * 48 + c4] = *(const uint4*)(A + (size_t)(bm0 + r) * GK + k0 + c4);
      *(uint4*)&Bs[r * 48 + c4] = *(const uint4*)(Wt + (size_t)(bn0 + r) * GK + k0 + c4);
    }
#endif
    if (k0 + 32 < GK) {  // global_prefetch_b8 for next K tiles
      __builtin_prefetch(A + (size_t)(bm0 + (tid >> 1)) * GK + k0 + 32, 0, 1);
      __builtin_prefetch(Wt + (size_t)(bn0 + (tid >> 1)) * GK + k0 + 32, 0, 1);
    }
#ifdef HAVE_ASYNC_LDS
    wait_async0();
#endif
    __syncthreads();

    v16h a[4], b[2];
    #pragma unroll
    for (int mt = 0; mt < 4; ++mt)
      a[mt] = frag_a(&As[(wm * 64 + mt * 16 + laneLo) * 48], hi ? 8 : 0);
    #pragma unroll
    for (int nt = 0; nt < 2; ++nt)
      b[nt] = frag_b(&Bs[(wn * 32 + nt * 16 + laneLo) * 48], hi ? 8 : 0);
    #pragma unroll
    for (int mt = 0; mt < 4; ++mt)
      #pragma unroll
      for (int nt = 0; nt < 2; ++nt)
        acc[mt][nt] = __builtin_amdgcn_wmma_f32_16x16x32_f16(
            false, a[mt], false, b[nt], (short)0, acc[mt][nt], false, false);
  }

  #pragma unroll
  for (int mt = 0; mt < 4; ++mt)
    #pragma unroll
    for (int nt = 0; nt < 2; ++nt)
      #pragma unroll
      for (int e = 0; e < 8; ++e) {
        int row = bm0 + wm * 64 + mt * 16 + e + hi * 8;
        int col = bn0 + wn * 32 + nt * 16 + laneLo;
        float val = acc[mt][nt][e] + bias[col];
        if (mode == 0) {
          outF32[(size_t)row * GN + col] = val;
        } else {
          int b_ = row / NS, n_ = row % NS;
          int h_ = col / HD, hd_ = col % HD;
          outSplit[((size_t)(b_ * 8 + h_) * NS + n_) * HD + hd_] = (_Float16)val;
        }
      }
}

// ---------------------------------------------------------------------------
// Focusing: per row (768 ch): softplus scale, relu+eps, cube, renorm to old L2.
// ---------------------------------------------------------------------------
__global__ __launch_bounds__(256) void focus_kernel(
    const float* __restrict__ qraw, const float* __restrict__ kraw,
    const float* __restrict__ scale_param, const float* __restrict__ pos_enc,
    _Float16* __restrict__ qs, _Float16* __restrict__ ks)
{
  const int r = blockIdx.x;
  const int tid = threadIdx.x;
  const int n = r % NS;
  float pq3[3], pk3[3];
  float s2q = 0.f, s6q = 0.f, s2k = 0.f, s6k = 0.f;
  #pragma unroll
  for (int i = 0; i < 3; ++i) {
    int c = tid + i * 256;
    float scale = log1pf(expf(scale_param[c]));   // softplus
    float qv = qraw[(size_t)r * GK + c];
    float kv = kraw[(size_t)r * GK + c] + pos_enc[(size_t)n * GK + c];
    float pq = (fmaxf(qv, 0.f) + 1e-6f) / scale;
    float pk = (fmaxf(kv, 0.f) + 1e-6f) / scale;
    s2q += pq * pq; s2k += pk * pk;
    float q3 = pq * pq * pq, k3 = pk * pk * pk;
    pq3[i] = q3; pk3[i] = k3;
    s6q += q3 * q3; s6k += k3 * k3;
  }
  #pragma unroll
  for (int o = 16; o > 0; o >>= 1) {   // wave32 reduction
    s2q += __shfl_xor(s2q, o, 32); s6q += __shfl_xor(s6q, o, 32);
    s2k += __shfl_xor(s2k, o, 32); s6k += __shfl_xor(s6k, o, 32);
  }
  __shared__ float red[8][4];
  if ((tid & 31) == 0) {
    int w = tid >> 5;
    red[w][0] = s2q; red[w][1] = s6q; red[w][2] = s2k; red[w][3] = s6k;
  }
  __syncthreads();
  float t2q = 0, t6q = 0, t2k = 0, t6k = 0;
  #pragma unroll
  for (int w = 0; w < 8; ++w) {
    t2q += red[w][0]; t6q += red[w][1]; t2k += red[w][2]; t6k += red[w][3];
  }
  float fq = sqrtf(t2q) / sqrtf(t6q);
  float fk = sqrtf(t2k) / sqrtf(t6k);
  int b_ = r / NS;
  #pragma unroll
  for (int i = 0; i < 3; ++i) {
    int c = tid + i * 256;
    int h_ = c / HD, hd_ = c % HD;
    size_t di = ((size_t)(b_ * 8 + h_) * NS + n) * HD + hd_;
    qs[di] = (_Float16)(pq3[i] * fq);
    ks[di] = (_Float16)(pk3[i] * fk);
  }
}

// ksum[bh][hd] = sum_n k[bh][n][hd]
__global__ __launch_bounds__(128) void ksum_kernel(
    const _Float16* __restrict__ ks, float* __restrict__ ksum)
{
  int bh = blockIdx.x, hd = threadIdx.x;
  if (hd >= HD) return;
  const _Float16* p = ks + (size_t)bh * NS * HD + hd;
  float s = 0.f;
  for (int nn = 0; nn < NS; ++nn) s += (float)p[nn * HD];
  ksum[bh * HD + hd] = s;
}

// z[bh][i] = 1/(q[bh][i]·ksum[bh] + 1e-6)
__global__ __launch_bounds__(224) void z_kernel(
    const _Float16* __restrict__ qs, const float* __restrict__ ksum,
    float* __restrict__ z)
{
  int bh = blockIdx.x, i = threadIdx.x;
  if (i >= NS) return;
  const _Float16* q = qs + ((size_t)bh * NS + i) * HD;
  const float* s = ksum + bh * HD;
  float d = 0.f;
  for (int c = 0; c < HD; ++c) d += (float)q[c] * s[c];
  z[bh * NS + i] = 1.f / (d + 1e-6f);
}

// ---------------------------------------------------------------------------
// kv[bh] = k^T v : M=96 (c), N=96 (d), K=196 (zero-padded). 4 waves, 48x48 each.
// ---------------------------------------------------------------------------
__global__ __launch_bounds__(128) void kv_kernel(
    const _Float16* __restrict__ ks, const _Float16* __restrict__ vs,
    _Float16* __restrict__ kv)
{
  __shared__ __align__(16) _Float16 As[96 * 48];
  __shared__ __align__(16) _Float16 Bs[96 * 48];
  const int bh = blockIdx.x;
  const int tid = threadIdx.x;
  const int lane = tid & 31, laneLo = lane & 15, hi = (lane >> 4) & 1;
  const int wave = tid >> 5;
  const int wm = wave & 1, wn = wave >> 1;
  const _Float16* kp = ks + (size_t)bh * NS * HD;
  const _Float16* vp = vs + (size_t)bh * NS * HD;

  v8f acc[3][3];
  #pragma unroll
  for (int i = 0; i < 3; ++i)
    #pragma unroll
    for (int j = 0; j < 3; ++j)
      #pragma unroll
      for (int e = 0; e < 8; ++e) acc[i][j][e] = 0.f;

  for (int j0 = 0; j0 < NS; j0 += 32) {
    if (j0) __syncthreads();
    for (int idx = tid; idx < 3072; idx += 128) {   // 32x96 chunk, transposed
      int j = idx / 96, c = idx % 96;
      int jj = j0 + j;
      _Float16 kval = (jj < NS) ? kp[(size_t)jj * HD + c] : (_Float16)0.f;
      _Float16 vval = (jj < NS) ? vp[(size_t)jj * HD + c] : (_Float16)0.f;
      As[c * 48 + j] = kval;   // A[m=c][k=j] = k^T
      Bs[c * 48 + j] = vval;   // Bs[n=d][k=j]
    }
    __syncthreads();
    v16h a[3], b[3];
    #pragma unroll
    for (int mt = 0; mt < 3; ++mt)
      a[mt] = frag_a(&As[(wm * 48 + mt * 16 + laneLo) * 48], hi ? 8 : 0);
    #pragma unroll
    for (int nt = 0; nt < 3; ++nt)
      b[nt] = frag_b(&Bs[(wn * 48 + nt * 16 + laneLo) * 48], hi ? 8 : 0);
    #pragma unroll
    for (int mt = 0; mt < 3; ++mt)
      #pragma unroll
      for (int nt = 0; nt < 3; ++nt)
        acc[mt][nt] = __builtin_amdgcn_wmma_f32_16x16x32_f16(
            false, a[mt], false, b[nt], (short)0, acc[mt][nt], false, false);
  }
  #pragma unroll
  for (int mt = 0; mt < 3; ++mt)
    #pragma unroll
    for (int nt = 0; nt < 3; ++nt)
      #pragma unroll
      for (int e = 0; e < 8; ++e) {
        int c = wm * 48 + mt * 16 + e + hi * 8;
        int d = wn * 48 + nt * 16 + laneLo;
        kv[(size_t)bh * (HD * HD) + c * HD + d] = (_Float16)acc[mt][nt][e];
      }
}

// ---------------------------------------------------------------------------
// Depthwise 5x5 SAME conv on v (14x14/channel) + bias -> conv f32 (merged layout)
// ---------------------------------------------------------------------------
__global__ __launch_bounds__(256) void dwc_kernel(
    const _Float16* __restrict__ vs, const float* __restrict__ w,
    const float* __restrict__ bias, float* __restrict__ conv)
{
  __shared__ float wsm[HD * 25];
  const int bh = blockIdx.x;
  for (int idx = threadIdx.x; idx < HD * 25; idx += 256) wsm[idx] = w[idx];
  __syncthreads();
  const int b_ = bh >> 3, h_ = bh & 7;
  const _Float16* vp = vs + (size_t)bh * NS * HD;
  for (int idx = threadIdx.x; idx < NS * HD; idx += 256) {
    int n = idx / HD, hd = idx % HD;
    int y = n / 14, x = n % 14;
    float s = bias[hd];
    #pragma unroll
    for (int ky = 0; ky < 5; ++ky) {
      int yy = y + ky - 2;
      if (yy < 0 || yy >= 14) continue;
      #pragma unroll
      for (int kx = 0; kx < 5; ++kx) {
        int xx = x + kx - 2;
        if (xx < 0 || xx >= 14) continue;
        s += (float)vp[(size_t)(yy * 14 + xx) * HD + hd] * wsm[hd * 25 + ky * 5 + kx];
      }
    }
    conv[((size_t)b_ * NS + n) * GN + h_ * HD + hd] = s;
  }
}

// ---------------------------------------------------------------------------
// out[bh] = (q @ kv)*z + conv, written merged-head f16 (feeds final GEMM).
// M=196 (pad 208), N=96, K=96. One block/bh, 8 waves sweep 13x6 tiles.
// ---------------------------------------------------------------------------
__global__ __launch_bounds__(256) void out_kernel(
    const _Float16* __restrict__ qs, const _Float16* __restrict__ kv,
    const float* __restrict__ z, const float* __restrict__ conv,
    _Float16* __restrict__ attn_h)
{
  __shared__ __align__(16) _Float16 As[208 * 96];   // stride 96 halves = 192B
  __shared__ __align__(16) _Float16 Bs[96 * 96];
  const int bh = blockIdx.x;
  const int tid = threadIdx.x;
  const int lane = tid & 31, laneLo = lane & 15, hi = (lane >> 4) & 1;
  const int wave = tid >> 5;
  const _Float16* qp = qs + (size_t)bh * NS * HD;
  const _Float16* kvp = kv + (size_t)bh * (HD * HD);

  for (int idx = tid; idx < 208 * 96; idx += 256) {
    int i = idx / 96, c = idx % 96;
    As[idx] = (i < NS) ? qp[(size_t)i * HD + c] : (_Float16)0.f;
  }
  for (int idx = tid; idx < 96 * 96; idx += 256) {
    int d = idx / 96, c = idx % 96;
    Bs[idx] = kvp[c * HD + d];                       // Bs[n=d][k=c]
  }
  __syncthreads();

  const int b_ = bh >> 3, h_ = bh & 7;
  for (int t = wave; t < 78; t += 8) {
    int mt = t / 6, nt = t % 6;
    v8f acc;
    #pragma unroll
    for (int e = 0; e < 8; ++e) acc[e] = 0.f;
    #pragma unroll
    for (int kt = 0; kt < 3; ++kt) {
      v16h a = frag_a(&As[(mt * 16 + laneLo) * 96], kt * 32 + (hi ? 8 : 0));
      v16h b = frag_b(&Bs[(nt * 16 + laneLo) * 96], kt * 32 + (hi ? 8 : 0));
      acc = __builtin_amdgcn_wmma_f32_16x16x32_f16(
          false, a, false, b, (short)0, acc, false, false);
    }
    #pragma unroll
    for (int e = 0; e < 8; ++e) {
      int i = mt * 16 + e + hi * 8;
      if (i < NS) {
        int d = nt * 16 + laneLo;
        size_t oi = ((size_t)b_ * NS + i) * GN + h_ * HD + d;
        attn_h[oi] = (_Float16)(acc[e] * z[bh * NS + i] + conv[oi]);
      }
    }
  }
}

// ---------------------------------------------------------------------------
extern "C" void kernel_launch(void* const* d_in, const int* in_sizes, int n_in,
                              void* d_out, int out_size, void* d_ws, size_t ws_size,
                              hipStream_t stream) {
  const float* x           = (const float*)d_in[0];
  const float* Wq          = (const float*)d_in[1];
  const float* bq          = (const float*)d_in[2];
  const float* Wk          = (const float*)d_in[3];
  const float* bk          = (const float*)d_in[4];
  const float* Wv          = (const float*)d_in[5];
  const float* bv          = (const float*)d_in[6];
  const float* pos_enc     = (const float*)d_in[7];
  const float* scale_param = (const float*)d_in[8];
  const float* dwc_w       = (const float*)d_in[9];
  const float* dwc_b       = (const float*)d_in[10];
  const float* Wp          = (const float*)d_in[11];
  const float* bp          = (const float*)d_in[12];
  float* out = (float*)d_out;

  const size_t MC = (size_t)GM * GK;           // 19,267,584 == BH*NS*HD
  // f32 region first (4B), then f16 region (all slices 16B aligned)
  float*    f_qraw = (float*)d_ws;
  float*    f_kraw = f_qraw + MC;
  float*    f_ksum = f_kraw + MC;
  float*    f_z    = f_ksum + (size_t)BH * HD;
  _Float16* h_x    = (_Float16*)(f_z + (size_t)BH * NS);
  _Float16* h_wt   = h_x + MC;                 // 4 transposed f16 weights
  _Float16* h_qs   = h_wt + 4 * WSZ;
  _Float16* h_ks   = h_qs + MC;
  _Float16* h_vs   = h_ks + MC;
  _Float16* h_kv   = h_vs + MC;
  float*    f_conv = f_qraw;                   // q raw dead after focus_kernel
  _Float16* h_attn = h_ks;                     // k split dead after kv_kernel

  dim3 g(GN / 128, GM / 128), blk(256);
  cvt_f16_kernel<<<MC / 1024, 256, 0, stream>>>(x, h_x);
  wtrans_kernel<<<dim3(12, 12, 4), 256, 0, stream>>>(Wq, Wk, Wv, Wp, h_wt);
  gemm768h_kernel<<<g, blk, 0, stream>>>(h_x, h_wt + 0 * WSZ, bq, f_qraw, nullptr, 0);
  gemm768h_kernel<<<g, blk, 0, stream>>>(h_x, h_wt + 1 * WSZ, bk, f_kraw, nullptr, 0);
  gemm768h_kernel<<<g, blk, 0, stream>>>(h_x, h_wt + 2 * WSZ, bv, nullptr, h_vs, 1);
  focus_kernel<<<GM, 256, 0, stream>>>(f_qraw, f_kraw, scale_param, pos_enc, h_qs, h_ks);
  ksum_kernel<<<BH, 128, 0, stream>>>(h_ks, f_ksum);
  z_kernel<<<BH, 224, 0, stream>>>(h_qs, f_ksum, f_z);
  kv_kernel<<<BH, 128, 0, stream>>>(h_ks, h_vs, h_kv);
  dwc_kernel<<<BH, 256, 0, stream>>>(h_vs, dwc_w, dwc_b, f_conv);
  out_kernel<<<BH, 256, 0, stream>>>(h_qs, h_kv, f_z, f_conv, h_attn);
  gemm768h_kernel<<<g, blk, 0, stream>>>(h_attn, h_wt + 3 * WSZ, bp, out, nullptr, 0);
}
